// encoder_32504312496827
// MI455X (gfx1250) — compile-verified, matrix operand order checked by reading, not run on hardware
//
#include <hip/hip_runtime.h>
#include <hip/hip_bf16.h>

// ---------------------------------------------------------------------------
// Problem constants (from reference)
// ---------------------------------------------------------------------------
#define BN_EPS 1e-5f

typedef __attribute__((ext_vector_type(16))) __bf16 v16bf;
typedef __attribute__((ext_vector_type(8)))  float  v8f;

// ---------------------------------------------------------------------------
// Kernel 0: zero a float buffer
// ---------------------------------------------------------------------------
__global__ void k_zero(float* __restrict__ p, int n) {
    int i = blockIdx.x * blockDim.x + threadIdx.x;
    if (i < n) p[i] = 0.0f;
}

// ---------------------------------------------------------------------------
// Kernel 1: pre-pack M matrices into bf16 WMMA-B-fragment tile layout.
//   M[(kp*IN + i), o] = kp<8 ? w_nn[(i*OUT+o)*8 + kp] : (kp==8 ? b_nn[i*OUT+o] : 0)
// Fragment layout: elem((kb,nb,lane,j)) = M[kb*32 + kmap(j,half)][nb*16 + (lane&15)]
//   kmap(j,half) = (j<8) ? half*8+j : 8+half*8+j   (ISA 16-bit A/B K striping)
// Flat index: ((kb*NB + nb)*32 + lane)*16 + j  -> 32 contiguous bytes per lane.
// ---------------------------------------------------------------------------
__device__ __forceinline__ float mval(const float* __restrict__ w,
                                      const float* __restrict__ b,
                                      int IN, int OUT, int flat, int o) {
    int kp = flat / IN;
    int i  = flat % IN;
    if (kp < 8) return w[(i * OUT + o) * 8 + kp];
    if (kp == 8) return b[i * OUT + o];
    return 0.0f;
}

__global__ void k_prep_bfrag(const float* __restrict__ w0, const float* __restrict__ b0,
                             const float* __restrict__ w1, const float* __restrict__ b1,
                             __bf16* __restrict__ f0, __bf16* __restrict__ f1) {
    // layer0: IN=16 OUT=32 KB=5 NB=2  -> 5*2*32*16 = 5120 elems
    // layer1: IN=32 OUT=64 KB=9 NB=4  -> 9*4*32*16 = 18432 elems
    const int n0 = 5 * 2 * 32 * 16;
    const int n1 = 9 * 4 * 32 * 16;
    for (int idx = blockIdx.x * blockDim.x + threadIdx.x; idx < n0 + n1;
         idx += gridDim.x * blockDim.x) {
        int IN, OUT, NB, lidx;
        const float *w, *b;
        __bf16* f;
        if (idx < n0) { IN = 16; OUT = 32; NB = 2; lidx = idx;       w = w0; b = b0; f = f0; }
        else          { IN = 32; OUT = 64; NB = 4; lidx = idx - n0;  w = w1; b = b1; f = f1; }
        const int j    = lidx & 15;
        const int lane = (lidx >> 4) & 31;
        const int tnb  = lidx >> 9;
        const int nb   = tnb % NB;
        const int kb   = tnb / NB;
        const int hlf  = lane >> 4;
        const int n    = lane & 15;
        const int kmap = (j < 8) ? (hlf * 8 + j) : (8 + hlf * 8 + j);
        const int flat = kb * 32 + kmap;
        const int o    = nb * 16 + n;
        f[lidx] = (__bf16)mval(w, b, IN, OUT, flat, o);
    }
}

// ---------------------------------------------------------------------------
// Kernel 2: agg[n][o] = bias[o] + sum_i x[n][i] * root[i][o]
// ---------------------------------------------------------------------------
template <int IN, int OUT>
__global__ void k_node_root(const float* __restrict__ x, const float* __restrict__ root,
                            const float* __restrict__ bias, float* __restrict__ agg, int N) {
    int idx = blockIdx.x * blockDim.x + threadIdx.x;
    if (idx >= N * OUT) return;
    const int n = idx / OUT;
    const int o = idx - n * OUT;
    float s = bias[o];
#pragma unroll
    for (int i = 0; i < IN; ++i) s += x[(size_t)n * IN + i] * root[i * OUT + o];
    agg[idx] = s;
}

// ---------------------------------------------------------------------------
// Kernel 3: fused NNConv edge path with WMMA.
//   Each wave owns a 16-edge tile. A (16 x 32k bf16) built from staged
//   x[src] outer u; B fragments streamed from prepacked global; C f32.
//   Scatter-add into agg[dst] with f32 global atomics.
// ---------------------------------------------------------------------------
template <int IN, int OUT, int KB, int NB>
__global__ __launch_bounds__(128) void k_nnconv_edge(
    const float* __restrict__ x, const int* __restrict__ eidx,
    const float* __restrict__ eattr, const __bf16* __restrict__ bfrag,
    float* __restrict__ agg, int E, int ntiles) {
    constexpr int WAVES = 4;
    __shared__ float xs[WAVES * 16 * IN];
    __shared__ float uu[WAVES * 16 * 12];
    __shared__ int   dl[WAVES * 16];

    const int wave = threadIdx.x >> 5;
    const int lane = threadIdx.x & 31;
    const int hlf  = lane >> 4;
    const int ln16 = lane & 15;
    float* xsw = xs + wave * 16 * IN;
    float* uuw = uu + wave * 16 * 12;
    int*   dlw = dl + wave * 16;

    const v16bf* bbase = reinterpret_cast<const v16bf*>(bfrag);
    __builtin_prefetch(bfrag, 0, 3);  // global_prefetch_b8: warm M tiles

    for (int t0 = blockIdx.x * WAVES; t0 < ntiles; t0 += gridDim.x * WAVES) {
        const int tile = t0 + wave;
        const bool act = tile < ntiles;

        // ---- stage 16 edges: x[src] (IN f32), u = [attr(8),1,0,0,0], dst ----
        if (act && lane < 16) {
            const int e = tile * 16 + lane;
            int dst = 0;
            if (e < E) {
                const int src = eidx[e];
                dst = eidx[E + e];
                const float4* xp = reinterpret_cast<const float4*>(x + (size_t)src * IN);
                float4* xd = reinterpret_cast<float4*>(xsw + lane * IN);
#pragma unroll
                for (int i = 0; i < IN / 4; ++i) xd[i] = xp[i];
                const float4* ap = reinterpret_cast<const float4*>(eattr + (size_t)e * 8);
                float4* ud = reinterpret_cast<float4*>(uuw + lane * 12);
                ud[0] = ap[0];
                ud[1] = ap[1];
                ud[2] = make_float4(1.0f, 0.0f, 0.0f, 0.0f);
            } else {
#pragma unroll
                for (int i = 0; i < IN; ++i) xsw[lane * IN + i] = 0.0f;
#pragma unroll
                for (int i = 0; i < 12; ++i) uuw[lane * 12 + i] = 0.0f;
            }
            dlw[lane] = dst;
        }
        __syncthreads();

        if (act) {
            v8f acc[NB];
#pragma unroll
            for (int nb = 0; nb < NB; ++nb) acc[nb] = (v8f){0, 0, 0, 0, 0, 0, 0, 0};

#pragma unroll
            for (int kb = 0; kb < KB; ++kb) {
                // Build A fragment row m = ln16; K striping per ISA table.
                v16bf a;
#pragma unroll
                for (int j = 0; j < 16; ++j) {
                    const int kmap = (j < 8) ? (hlf * 8 + j) : (8 + hlf * 8 + j);
                    const int flat = kb * 32 + kmap;
                    const int kp   = flat / IN;  // IN is power of two -> shift
                    const int i    = flat % IN;
                    a[j] = (__bf16)(xsw[ln16 * IN + i] * uuw[ln16 * 12 + kp]);
                }
#pragma unroll
                for (int nb = 0; nb < NB; ++nb) {
                    v16bf b = bbase[(kb * NB + nb) * 32 + lane];
                    acc[nb] = __builtin_amdgcn_wmma_f32_16x16x32_bf16(
                        false, a, false, b, (short)0, acc[nb], false, false);
                }
            }

            // ---- scatter-add: C/D layout: reg r -> M = r + 8*half, N = ln16 ----
#pragma unroll
            for (int nb = 0; nb < NB; ++nb) {
                const int o = nb * 16 + ln16;
#pragma unroll
                for (int r = 0; r < 8; ++r) {
                    const int m = r + 8 * hlf;
                    float* dstp = &agg[(size_t)dlw[m] * OUT + o];
                    __hip_atomic_fetch_add(dstp, acc[nb][r], __ATOMIC_RELAXED,
                                           __HIP_MEMORY_SCOPE_AGENT);
                }
            }
        }
        __syncthreads();
    }
}

// ---------------------------------------------------------------------------
// Kernel 4: BN stats (block per channel): scale/shift from biased mean/var.
// ---------------------------------------------------------------------------
__global__ __launch_bounds__(256) void k_bn_stats(const float* __restrict__ h,
                                                  const float* __restrict__ gamma,
                                                  const float* __restrict__ beta,
                                                  float* __restrict__ scale,
                                                  float* __restrict__ shift, int N, int C) {
    const int c = blockIdx.x;
    __shared__ float s1[256], s2[256];
    float a = 0.0f, b = 0.0f;
    for (int n = threadIdx.x; n < N; n += 256) {
        float v = h[(size_t)n * C + c];
        a += v;
        b += v * v;
    }
    s1[threadIdx.x] = a;
    s2[threadIdx.x] = b;
    __syncthreads();
    for (int st = 128; st > 0; st >>= 1) {
        if (threadIdx.x < st) {
            s1[threadIdx.x] += s1[threadIdx.x + st];
            s2[threadIdx.x] += s2[threadIdx.x + st];
        }
        __syncthreads();
    }
    if (threadIdx.x == 0) {
        const float inv_n = 1.0f / (float)N;
        const float mu    = s1[0] * inv_n;
        const float var   = s2[0] * inv_n - mu * mu;
        const float sc    = gamma[c] * rsqrtf(var + BN_EPS);
        scale[c] = sc;
        shift[c] = beta[c] - mu * sc;
    }
}

// ---------------------------------------------------------------------------
// Kernel 5: in-place BN + ReLU
// ---------------------------------------------------------------------------
__global__ void k_bn_relu(float* __restrict__ h, const float* __restrict__ scale,
                          const float* __restrict__ shift, int N, int C) {
    int idx = blockIdx.x * blockDim.x + threadIdx.x;
    if (idx >= N * C) return;
    const int c = idx % C;
    const float v = h[idx] * scale[c] + shift[c];
    h[idx] = fmaxf(v, 0.0f);
}

// ---------------------------------------------------------------------------
// Kernel 6: heads + gate + segment-sum pooling. Block per node (128 thr).
// ---------------------------------------------------------------------------
__global__ __launch_bounds__(128) void k_head(const float* __restrict__ h,
                                              const float* __restrict__ sw,
                                              const float* __restrict__ sb,
                                              const float* __restrict__ tw,
                                              const float* __restrict__ tb,
                                              const int* __restrict__ batch,
                                              float* __restrict__ out, int N) {
    const int n = blockIdx.x;
    if (n >= N) return;
    __shared__ float hs[64];
    if (threadIdx.x < 64) hs[threadIdx.x] = h[(size_t)n * 64 + threadIdx.x];
    __syncthreads();
    const int o = threadIdx.x;  // 0..127
    float s = sb[o], t = tb[o];
#pragma unroll 8
    for (int i = 0; i < 64; ++i) {
        const float hv = hs[i];
        s += hv * sw[o * 64 + i];
        t += hv * tw[o * 64 + i];
    }
    s = fminf(fmaxf(s, -30.0f), 30.0f);
    const float f = (1.0f / (1.0f + expf(s))) * tanhf(t);
    float* dstp = &out[(size_t)batch[n] * 128 + o];
    __hip_atomic_fetch_add(dstp, f, __ATOMIC_RELAXED, __HIP_MEMORY_SCOPE_AGENT);
}

// ---------------------------------------------------------------------------
// Host launcher
// ---------------------------------------------------------------------------
static inline size_t align_up(size_t v, size_t a) { return (v + a - 1) & ~(a - 1); }

extern "C" void kernel_launch(void* const* d_in, const int* in_sizes, int n_in,
                              void* d_out, int out_size, void* d_ws, size_t ws_size,
                              hipStream_t stream) {
    const float* x        = (const float*)d_in[0];
    const int*   eidx     = (const int*)d_in[1];
    const float* eattr    = (const float*)d_in[2];
    const int*   batch    = (const int*)d_in[3];
    // d_in[4] edge_batch: unused by reference
    const float* w_nn0 = (const float*)d_in[5];
    const float* b_nn0 = (const float*)d_in[6];
    const float* root0 = (const float*)d_in[7];
    const float* bias0 = (const float*)d_in[8];
    const float* gam0  = (const float*)d_in[9];
    const float* bet0  = (const float*)d_in[10];
    const float* w_nn1 = (const float*)d_in[11];
    const float* b_nn1 = (const float*)d_in[12];
    const float* root1 = (const float*)d_in[13];
    const float* bias1 = (const float*)d_in[14];
    const float* gam1  = (const float*)d_in[15];
    const float* bet1  = (const float*)d_in[16];
    const float* s_w   = (const float*)d_in[17];
    const float* s_b   = (const float*)d_in[18];
    const float* t_w   = (const float*)d_in[19];
    const float* t_b   = (const float*)d_in[20];
    float* out = (float*)d_out;

    const int N = in_sizes[3];      // batch has N entries
    const int E = in_sizes[4];      // edge_batch has E entries

    // Workspace carve-out
    char*  base = (char*)d_ws;
    size_t off  = 0;
    float* agg0 = (float*)(base + off); off = align_up(off + (size_t)N * 32 * 4, 256);
    float* agg1 = (float*)(base + off); off = align_up(off + (size_t)N * 64 * 4, 256);
    __bf16* bf0 = (__bf16*)(base + off); off = align_up(off + 5120 * 2, 256);
    __bf16* bf1 = (__bf16*)(base + off); off = align_up(off + 18432 * 2, 256);
    float* sc0  = (float*)(base + off); off = align_up(off + 32 * 4, 256);
    float* sh0  = (float*)(base + off); off = align_up(off + 32 * 4, 256);
    float* sc1  = (float*)(base + off); off = align_up(off + 64 * 4, 256);
    float* sh1  = (float*)(base + off); off = align_up(off + 64 * 4, 256);
    (void)ws_size;

    const int ntiles = (E + 15) / 16;
    const int edge_blocks = (ntiles + 3) / 4;

    // 0) zero output accumulator
    k_zero<<<(out_size + 255) / 256, 256, 0, stream>>>(out, out_size);

    // 1) pre-pack bf16 B fragments for both layers
    k_prep_bfrag<<<92, 256, 0, stream>>>(w_nn0, b_nn0, w_nn1, b_nn1, bf0, bf1);

    // ---- layer 0 ----
    k_node_root<16, 32><<<(N * 32 + 255) / 256, 256, 0, stream>>>(x, root0, bias0, agg0, N);
    k_nnconv_edge<16, 32, 5, 2><<<edge_blocks, 128, 0, stream>>>(x, eidx, eattr, bf0, agg0, E, ntiles);
    k_bn_stats<<<32, 256, 0, stream>>>(agg0, gam0, bet0, sc0, sh0, N, 32);
    k_bn_relu<<<(N * 32 + 255) / 256, 256, 0, stream>>>(agg0, sc0, sh0, N, 32);

    // ---- layer 1 (input = agg0) ----
    k_node_root<32, 64><<<(N * 64 + 255) / 256, 256, 0, stream>>>(agg0, root1, bias1, agg1, N);
    k_nnconv_edge<32, 64, 9, 4><<<edge_blocks, 128, 0, stream>>>(agg0, eidx, eattr, bf1, agg1, E, ntiles);
    k_bn_stats<<<64, 256, 0, stream>>>(agg1, gam1, bet1, sc1, sh1, N, 64);
    k_bn_relu<<<(N * 64 + 255) / 256, 256, 0, stream>>>(agg1, sc1, sh1, N, 64);

    // ---- heads + pooling ----
    k_head<<<N, 128, 0, stream>>>(agg1, s_w, s_b, t_w, t_b, batch, out, N);
}
